// Attention_661424963678
// MI455X (gfx1250) — matrix operand order repaired and checked by reading, not compiled, not run on hardware
//
#include <hip/hip_runtime.h>
#include <hip/hip_bf16.h>

typedef __attribute__((ext_vector_type(16))) _Float16 v16h;
typedef __attribute__((ext_vector_type(8)))  float    v8f;

#define WIN    25
#define N_TOK  625
#define N_PAD  640     // 625 padded to 40 tiles of 16
#define DMODEL 256
#define NHEAD  8
#define DHEAD  32
#define BATCH  32
#define RT_N   40      // row tiles (N_PAD/16)
#define KK_N   8       // K chunks of 32 over DMODEL

// ---------------------------------------------------------------------------
// WMMA fragment maps, gfx1250 wave32 (CDNA5 ISA §7.12.2).
// All operands are PRE-SWIZZLED in memory into per-lane fragment order:
// 16 contiguous halves (32 B) per lane -> one aligned v16h load (2x b128).
//
// A (16xK32 f16):  lane = rowA + 16*((K>>3)&1),  slot = (K&7) + 8*(K>>4)
// B (K32x16 f16):  lane = col  + 16*((K>>3)&1),  slot = (K&7) + 8*(K>>4)
// C/D (16x16 f32): lane = col  + 16*rowhalf,     vgpr r -> row = rowhalf*8 + r
// ---------------------------------------------------------------------------
__device__ __host__ __forceinline__ int a_lane(int rowA, int K) { return rowA + 16 * ((K >> 3) & 1); }
__device__ __host__ __forceinline__ int a_slot(int K)           { return (K & 7) + 8 * (K >> 4); }
__device__ __host__ __forceinline__ int b_lane(int col, int K)  { return col + 16 * ((K >> 3) & 1); }
__device__ __host__ __forceinline__ int b_slot(int K)           { return (K & 7) + 8 * (K >> 4); }

__device__ __forceinline__ v8f wmma_f16(v16h a, v16h b, v8f c) {
    return __builtin_amdgcn_wmma_f32_16x16x32_f16(false, a, false, b, (short)0, c, false, false);
}

// ---------------------------------------------------------------------------
// Pack x (b, d, 625) -> A-fragments xsw[b][rt][kk][lane][slot], zero-padded.
// Coalesced fp32 reads (token index fastest), scattered 2B fragment writes.
// ---------------------------------------------------------------------------
__global__ void pack_x_kernel(const float* __restrict__ x, _Float16* __restrict__ xsw) {
    const long total = (long)BATCH * DMODEL * N_PAD;
    for (long t = (long)blockIdx.x * blockDim.x + threadIdx.x; t < total;
         t += (long)gridDim.x * blockDim.x) {
        int  n  = (int)(t % N_PAD);
        long r  = t / N_PAD;
        int  dd = (int)(r % DMODEL);
        int  b  = (int)(r / DMODEL);
        float v = (n < N_TOK) ? x[((long)b * DMODEL + dd) * N_TOK + n] : 0.0f;
        int rt = n >> 4, rowA = n & 15, kk = dd >> 5, K = dd & 31;
        long idx = ((((long)b * RT_N + rt) * KK_N + kk) * 32 + a_lane(rowA, K)) * 16 + a_slot(K);
        xsw[idx] = (_Float16)v;
    }
}

// ---------------------------------------------------------------------------
// Pack weights -> B-fragments.
// Wqkvsw[kk][ct(48)][lane][slot], Woutsw[kk][ct(16)][lane][slot]
// ---------------------------------------------------------------------------
__global__ void pack_w_kernel(const float* __restrict__ Wqkv, const float* __restrict__ Wout,
                              _Float16* __restrict__ Wqkvsw, _Float16* __restrict__ Woutsw) {
    const int t1 = DMODEL * 3 * DMODEL;
    const int t2 = DMODEL * DMODEL;
    for (int t = blockIdx.x * blockDim.x + threadIdx.x; t < t1 + t2;
         t += gridDim.x * blockDim.x) {
        if (t < t1) {
            int c = t % (3 * DMODEL), r = t / (3 * DMODEL);
            int kk = r >> 5, K = r & 31, ct = c >> 4, colB = c & 15;
            Wqkvsw[((kk * 48 + ct) * 32 + b_lane(colB, K)) * 16 + b_slot(K)] = (_Float16)Wqkv[t];
        } else {
            int t0 = t - t1;
            int c = t0 % DMODEL, r = t0 / DMODEL;
            int kk = r >> 5, K = r & 31, ct = c >> 4, colB = c & 15;
            Woutsw[((kk * 16 + ct) * 32 + b_lane(colB, K)) * 16 + b_slot(K)] = (_Float16)Wout[t0];
        }
    }
}

// bias[h][i][j] = rel_emb[((iy-jy+24)*49 + (ix-jx+24)) * NHEAD + h]   (batch-invariant)
__global__ void make_bias_kernel(const float* __restrict__ rel, float* __restrict__ bias) {
    const long NN = (long)N_TOK * N_TOK;
    const long total = NHEAD * NN;
    for (long t = (long)blockIdx.x * blockDim.x + threadIdx.x; t < total;
         t += (long)gridDim.x * blockDim.x) {
        int  h   = (int)(t / NN);
        long rem = t % NN;
        int  i   = (int)(rem / N_TOK);
        int  j   = (int)(rem % N_TOK);
        int iy = i / WIN, ix = i % WIN, jy = j / WIN, jx = j % WIN;
        int r = (iy - jy + WIN - 1) * (2 * WIN - 1) + (ix - jx + WIN - 1);
        bias[t] = rel[r * NHEAD + h];
    }
}

// ---------------------------------------------------------------------------
// QKV projection: per batch (640x256)@(256x768). One wave computes a 16x64
// slab (4 accumulators, A fragment reused 4x). Scatters q/k/v directly into
// the fragment layouts the attention kernel consumes:
//   qsw[bh][rt][lane][slot]          (A-fragments, pre-scaled)
//   ksw[bh][jt][lane][slot]          (B-fragments of K^T, K-dim = dh)
//   vsw[bh][kt][ct2][lane][slot]     (B-fragments of V,  K-dim = token)
// ---------------------------------------------------------------------------
__global__ __launch_bounds__(32) void qkv_gemm_kernel(
    const _Float16* __restrict__ xsw, const _Float16* __restrict__ Wsw,
    _Float16* __restrict__ qsw, _Float16* __restrict__ ksw, _Float16* __restrict__ vsw) {
    const int lane = threadIdx.x;
    const int ct4 = blockIdx.x;  // 0..11 (768 / 64)
    const int rt  = blockIdx.y;  // 0..39
    const int b   = blockIdx.z;

    const v16h* Af = (const v16h*)xsw + ((long)b * RT_N + rt) * KK_N * 32;
    const v16h* Bf = (const v16h*)Wsw;
    v8f acc[4] = {{}, {}, {}, {}};
#pragma unroll
    for (int kk = 0; kk < KK_N; ++kk) {
        v16h a = Af[kk * 32 + lane];
        if (kk + 1 < KK_N)
            __builtin_prefetch(&Bf[((kk + 1) * 48 + ct4 * 4) * 32 + lane], 0, 0);
#pragma unroll
        for (int c = 0; c < 4; ++c) {
            v16h bm = Bf[(kk * 48 + ct4 * 4 + c) * 32 + lane];
            acc[c] = wmma_f16(a, bm, acc[c]);
        }
    }

    const int lo = lane & 15, hi = lane >> 4;
    const float scale = 0.1767766953f;  // 32^-0.5
#pragma unroll
    for (int c = 0; c < 4; ++c) {
        int col = (ct4 * 4 + c) * 16 + lo;
#pragma unroll
        for (int r = 0; r < 8; ++r) {
            int rl = hi * 8 + r;          // row within tile
            float v = acc[c][r];
            if (col < DMODEL) {           // ---- Q ----
                int head = col >> 5, dh = col & 31;
                long bh = (long)b * NHEAD + head;
                qsw[((bh * RT_N + rt) * 32 + a_lane(rl, dh)) * 16 + a_slot(dh)] =
                    (_Float16)(v * scale);
            } else if (col < 2 * DMODEL) {  // ---- K^T ----
                int c2 = col - DMODEL; int head = c2 >> 5, dh = c2 & 31;
                long bh = (long)b * NHEAD + head;
                ksw[((bh * RT_N + rt) * 32 + b_lane(rl, dh)) * 16 + b_slot(dh)] = (_Float16)v;
            } else {                        // ---- V ----
                int c2 = col - 2 * DMODEL; int head = c2 >> 5, dh = c2 & 31;
                long bh = (long)b * NHEAD + head;
                int tok = rt * 16 + rl;
                int kt = tok >> 5, kv = tok & 31, ct2 = dh >> 4, colB = dh & 15;
                vsw[(((bh * 20 + kt) * 2 + ct2) * 32 + b_lane(colB, kv)) * 16 + b_slot(kv)] =
                    (_Float16)v;
            }
        }
    }
}

// ---------------------------------------------------------------------------
// Fused attention: one wave per (batch, head, 16-row query slab).
// QK^T (40 WMMAs) -> bias+mask -> f32 logit fragments in LDS (v8f stores) ->
// softmax -> P written as A-fragments in LDS -> P@V (40 WMMAs) -> normalized
// f16 A-fragments for the output projection. sim never touches HBM.
// ---------------------------------------------------------------------------
__global__ __launch_bounds__(32) void attn_kernel(
    const _Float16* __restrict__ qsw, const _Float16* __restrict__ ksw,
    const _Float16* __restrict__ vsw, const float* __restrict__ bias,
    _Float16* __restrict__ asw) {
    __shared__ __align__(32) float    sim[16 * N_PAD];   // 40 KB, fragment order
    __shared__ __align__(32) _Float16 patt[16 * N_PAD];  // 20 KB, A-fragment order
    __shared__ float rinv[16];

    const int lane = threadIdx.x;
    const int rt = blockIdx.x;   // 0..39
    const int h  = blockIdx.y;
    const int b  = blockIdx.z;
    const int lo = lane & 15, hi = lane >> 4;

    const long bh = (long)b * NHEAD + h;
    const v16h qa = ((const v16h*)qsw)[(bh * RT_N + rt) * 32 + lane];
    const v16h* Kf = (const v16h*)ksw + bh * RT_N * 32;
    const v16h* Vf = (const v16h*)vsw + bh * 20 * 2 * 32;
    const float* hbias = bias + (long)h * N_TOK * N_TOK;

    // --- logits: 40 key tiles; D-fragment (v8f) stored contiguously in LDS ---
    for (int jt = 0; jt < RT_N; ++jt) {
        v16h bk = Kf[jt * 32 + lane];
        v8f s = {};
        s = wmma_f16(qa, bk, s);
        int colg = jt * 16 + lo;
#pragma unroll
        for (int r = 0; r < 8; ++r) {
            int rowg = rt * 16 + hi * 8 + r;
            float val = s[r];
            if (colg < N_TOK) {
                if (rowg < N_TOK) val += hbias[(long)rowg * N_TOK + colg];
            } else {
                val = -3.0e38f;  // padded key columns
            }
            s[r] = val;
        }
        *(v8f*)&sim[(jt * 32 + lane) * 8] = s;   // 2x ds_store_b128
    }
    __syncthreads();

    // --- softmax: lane handles row = lo over column half = hi ---
    // sim element (row, c) lives at ((c>>4)*32 + (c&15) + 16*(row>>3))*8 + (row&7)
    {
        const int row = lo;
        const int lgrp = 16 * (row >> 3), vr = row & 7;
        float m = -3.0e38f;
        for (int c = hi * (N_PAD / 2); c < (hi + 1) * (N_PAD / 2); ++c)
            m = fmaxf(m, sim[(((c >> 4) * 32) + (c & 15) + lgrp) * 8 + vr]);
        m = fmaxf(m, __shfl_xor(m, 16));
        float sum = 0.0f;
        for (int c = hi * (N_PAD / 2); c < (hi + 1) * (N_PAD / 2); ++c) {
            float p = __expf(sim[(((c >> 4) * 32) + (c & 15) + lgrp) * 8 + vr] - m);
            int kt = c >> 5, K = c & 31;
            patt[(kt * 32 + a_lane(row, K)) * 16 + a_slot(K)] = (_Float16)p;
            sum += p;
        }
        sum += __shfl_xor(sum, 16);
        if (hi == 0) rinv[row] = 1.0f / sum;
    }
    __syncthreads();

    // --- P @ V: 20 K-chunks of 32; P fragments are single aligned LDS loads ---
    v8f o0 = {}, o1 = {};
    for (int kt = 0; kt < N_PAD / 32; ++kt) {
        v16h aa = ((const v16h*)patt)[kt * 32 + lane];
        v16h b0 = Vf[(kt * 2 + 0) * 32 + lane];
        v16h b1 = Vf[(kt * 2 + 1) * 32 + lane];
        o0 = wmma_f16(aa, b0, o0);
        o1 = wmma_f16(aa, b1, o1);
    }

    // --- normalize + store as A-fragments for the output projection ---
    // channel c = h*32 + dh  ->  kk = h, K = dh (head-major concat)
#pragma unroll
    for (int r = 0; r < 8; ++r) {
        int rl = hi * 8 + r;
        float inv = rinv[rl];
        long base = (((long)b * RT_N + rt) * KK_N + h) * 32;
        int d0 = lo, d1 = 16 + lo;
        asw[(base + a_lane(rl, d0)) * 16 + a_slot(d0)] = (_Float16)(o0[r] * inv);
        asw[(base + a_lane(rl, d1)) * 16 + a_slot(d1)] = (_Float16)(o1[r] * inv);
    }
}

// ---------------------------------------------------------------------------
// Output projection: per batch (640x256)@(256x256), 16x64 slab per wave;
// stores fp32 transposed directly into (b, d, H, W), dropping padded rows.
// ---------------------------------------------------------------------------
__global__ __launch_bounds__(32) void out_gemm_kernel(
    const _Float16* __restrict__ asw, const _Float16* __restrict__ Wsw,
    float* __restrict__ out) {
    const int lane = threadIdx.x;
    const int ct4 = blockIdx.x;  // 0..3 (256 / 64)
    const int rt  = blockIdx.y;  // 0..39
    const int b   = blockIdx.z;

    const v16h* Af = (const v16h*)asw + ((long)b * RT_N + rt) * KK_N * 32;
    const v16h* Bf = (const v16h*)Wsw;
    v8f acc[4] = {{}, {}, {}, {}};
#pragma unroll
    for (int kk = 0; kk < KK_N; ++kk) {
        v16h a = Af[kk * 32 + lane];
        if (kk + 1 < KK_N)
            __builtin_prefetch(&Bf[((kk + 1) * 16 + ct4 * 4) * 32 + lane], 0, 0);
#pragma unroll
        for (int c = 0; c < 4; ++c) {
            v16h bm = Bf[(kk * 16 + ct4 * 4 + c) * 32 + lane];
            acc[c] = wmma_f16(a, bm, acc[c]);
        }
    }

    const int lo = lane & 15, hi = lane >> 4;
#pragma unroll
    for (int c = 0; c < 4; ++c) {
        int col = (ct4 * 4 + c) * 16 + lo;
#pragma unroll
        for (int r = 0; r < 8; ++r) {
            int row = rt * 16 + hi * 8 + r;
            if (row < N_TOK)
                out[((long)b * DMODEL + col) * N_TOK + row] = acc[c][r];
        }
    }
}

// ---------------------------------------------------------------------------
extern "C" void kernel_launch(void* const* d_in, const int* in_sizes, int n_in,
                              void* d_out, int out_size, void* d_ws, size_t ws_size,
                              hipStream_t stream) {
    const float* x    = (const float*)d_in[0];  // (32, 256, 25, 25)
    const float* Wqkv = (const float*)d_in[1];  // (256, 768)
    const float* Wout = (const float*)d_in[2];  // (256, 256)
    const float* rel  = (const float*)d_in[3];  // (2401, 8)
    float* out = (float*)d_out;                 // (32, 256, 25, 25)

    char* ws = (char*)d_ws;
    size_t off = 0;
    auto carve = [&](size_t bytes) {
        void* p = ws + off;
        off += (bytes + 255) & ~size_t(255);
        return p;
    };
    _Float16* xsw    = (_Float16*)carve(sizeof(_Float16) * (size_t)BATCH * RT_N * KK_N * 512);
    _Float16* Wqkvsw = (_Float16*)carve(sizeof(_Float16) * (size_t)KK_N * 48 * 512);
    _Float16* Woutsw = (_Float16*)carve(sizeof(_Float16) * (size_t)KK_N * 16 * 512);
    float*    bias   = (float*)   carve(sizeof(float) * (size_t)NHEAD * N_TOK * N_TOK);
    _Float16* qsw    = (_Float16*)carve(sizeof(_Float16) * (size_t)BATCH * NHEAD * RT_N * 512);
    _Float16* ksw    = (_Float16*)carve(sizeof(_Float16) * (size_t)BATCH * NHEAD * RT_N * 512);
    _Float16* vsw    = (_Float16*)carve(sizeof(_Float16) * (size_t)BATCH * NHEAD * 20 * 2 * 512);
    _Float16* asw    = (_Float16*)carve(sizeof(_Float16) * (size_t)BATCH * RT_N * KK_N * 512);
    (void)in_sizes; (void)n_in; (void)out_size; (void)ws_size;

    pack_x_kernel   <<<dim3(1024), 256, 0, stream>>>(x, xsw);
    pack_w_kernel   <<<dim3(256),  256, 0, stream>>>(Wqkv, Wout, Wqkvsw, Woutsw);
    make_bias_kernel<<<dim3(2048), 256, 0, stream>>>(rel, bias);

    qkv_gemm_kernel<<<dim3(12, RT_N, BATCH), 32, 0, stream>>>(xsw, Wqkvsw, qsw, ksw, vsw);
    attn_kernel    <<<dim3(RT_N, NHEAD, BATCH), 32, 0, stream>>>(qsw, ksw, vsw, bias, asw);
    out_gemm_kernel<<<dim3(4, RT_N, BATCH), 32, 0, stream>>>(asw, Woutsw, out);
}
